// HGTConv_55190329754179
// MI455X (gfx1250) — compile-verified, matrix operand order checked by reading, not run on hardware
//
#include <hip/hip_runtime.h>
#include <math.h>

// ---------------- problem constants ----------------
#define N_NODES 50000
#define N_EDGES 500000
#define C       128      // OUT_C == IN_C
#define CA      256      // 2*IN_C (augmented input)
#define H       8
#define DH      16

typedef __attribute__((ext_vector_type(16))) _Float16 v16h;
typedef __attribute__((ext_vector_type(8)))  _Float16 v8h;
typedef __attribute__((ext_vector_type(8)))  float    v8f;

// ---- ordered-uint encoding for float atomic max (monotone for all reals) ----
static __device__ __forceinline__ unsigned f2ord(float f) {
  unsigned u = __float_as_uint(f);
  return (u & 0x80000000u) ? ~u : (u | 0x80000000u);
}
static __device__ __forceinline__ float ord2f(unsigned u) {
  return (u & 0x80000000u) ? __uint_as_float(u & 0x7fffffffu) : __uint_as_float(~u);
}

static __device__ __forceinline__ void atomAddF(float* p, float v) {
  __hip_atomic_fetch_add(p, v, __ATOMIC_RELAXED, __HIP_MEMORY_SCOPE_AGENT);
}

// ---------------- WMMA fragment loaders ----------------
// CDNA5 16-bit A-matrix 16x32 layout (ISA 7.12.2): lane L holds row m = L%16;
// lanes 0-15: halves 0-7 = K k0..k0+7, halves 8-15 = K k0+16..k0+23;
// lanes 16-31: same +8.  => two contiguous 16B loads per lane.
// B (32x16) is fed from pre-transposed weights with the same striping.
static __device__ __forceinline__ v16h load_frag_f16(const _Float16* base, int ld, int k0) {
  const int lane = threadIdx.x & 31;
  const int m    = lane & 15;
  const int kb   = k0 + ((lane >> 4) << 3);
  const _Float16* p = base + m * ld + kb;
  v8h lo = *(const v8h*)(p);
  v8h hi = *(const v8h*)(p + 16);
  v16h r;
#pragma unroll
  for (int i = 0; i < 8; ++i) { r[i] = lo[i]; r[i + 8] = hi[i]; }
  return r;
}

static __device__ __forceinline__ v16h load_frag_f32cvt(const float* base, int ld, int k0) {
  const int lane = threadIdx.x & 31;
  const int m    = lane & 15;
  const int kb   = k0 + ((lane >> 4) << 3);
  const float* p = base + m * ld + kb;
  v16h r;
#pragma unroll
  for (int i = 0; i < 8; ++i) { r[i] = (_Float16)p[i]; r[i + 8] = (_Float16)p[i + 16]; }
  return r;
}

// ---------------- kernel 0: weight transpose + f16 convert ----------------
// WT1[n][k] (n in 0..511, k in 0..255) = {Wk,Wq,Wv,Wskip}[k][n%128]
// WoT[n][k] (128x128)                  = Wo[k][n]
__global__ void prep_weights(const float* __restrict__ Wk, const float* __restrict__ Wq,
                             const float* __restrict__ Wv, const float* __restrict__ Wskip,
                             const float* __restrict__ Wo,
                             _Float16* __restrict__ WT1, _Float16* __restrict__ WoT) {
  int t = blockIdx.x * 256 + threadIdx.x;
  if (t < 512 * 256) {
    int n = t >> 8, k = t & 255;
    const float* W = (n < 128) ? Wk : (n < 256) ? Wq : (n < 384) ? Wv : Wskip;
    WT1[n * 256 + k] = (_Float16)W[k * 128 + (n & 127)];
  } else {
    int u = t - 512 * 256;
    if (u < 128 * 128) {
      int n = u >> 7, k = u & 127;
      WoT[n * 128 + k] = (_Float16)Wo[k * 128 + n];
    }
  }
}

// ---------------- kernel 1: build x_aug (f16) ----------------
__global__ void build_xaug(const float* __restrict__ x, const int* __restrict__ ntype,
                           const float* __restrict__ type_emb, _Float16* __restrict__ xaug) {
  int t = blockIdx.x * 256 + threadIdx.x;     // N*256 threads exactly
  int n = t >> 8, c = t & 255;
  float v = (c < 128) ? x[n * 128 + c] : type_emb[ntype[n] * 128 + (c - 128)];
  xaug[t] = (_Float16)v;
}

// ---------------- kernel 2: fused K/Q/V/skip GEMM  [N,256]x[256,512] ----------------
// grid (N/16, 4), block 256 (8 waves). Wave w of blockIdx.y handles colTile = y*8+w.
__global__ void gemm_qkvs(const _Float16* __restrict__ xaug, const _Float16* __restrict__ WT1,
                          float* __restrict__ Qb, float* __restrict__ Kb,
                          float* __restrict__ Vb, float* __restrict__ Sb) {
  const int wave    = threadIdx.x >> 5;
  const int colTile = blockIdx.y * 8 + wave;      // 0..31
  const int row0    = blockIdx.x * 16;
  const _Float16* A = xaug + (size_t)row0 * CA;
  const _Float16* B = WT1 + (size_t)colTile * 16 * CA;
  v8f c = {};
#pragma unroll
  for (int k0 = 0; k0 < CA; k0 += 32) {
    v16h a = load_frag_f16(A, CA, k0);
    v16h b = load_frag_f16(B, CA, k0);
    c = __builtin_amdgcn_wmma_f32_16x16x32_f16(false, a, false, b, (short)0, c, false, false);
  }
  float* outb = (colTile < 8) ? Kb : (colTile < 16) ? Qb : (colTile < 24) ? Vb : Sb;
  const int col0 = (colTile & 7) * 16;
  const int lane = threadIdx.x & 31;
  const int n    = lane & 15;
  const int mb   = (lane >> 4) << 3;              // C/D layout: vgpr v -> row mb+v
#pragma unroll
  for (int v = 0; v < 8; ++v)
    outb[(size_t)(row0 + mb + v) * C + col0 + n] = c[v];
}

// ---------------- kernel 3: per-edge attention logits + global max ----------------
__global__ void attn_dot(const float* __restrict__ Qb, const float* __restrict__ Kb,
                         const int* __restrict__ esrc, const int* __restrict__ edst,
                         const int* __restrict__ erel, const float* __restrict__ rel_scale,
                         float* __restrict__ attn, unsigned* __restrict__ gmax) {
  int t = blockIdx.x * 256 + threadIdx.x;         // E*H threads exactly
  int e = t >> 3, h = t & 7;
  int s = esrc[e], d = edst[e];
  const float4* q = (const float4*)(Qb + (size_t)d * C + h * DH);
  const float4* k = (const float4*)(Kb + (size_t)s * C + h * DH);
  float acc = 0.f;
#pragma unroll
  for (int i = 0; i < 4; ++i) {
    float4 a = q[i], b = k[i];
    acc += a.x * b.x + a.y * b.y + a.z * b.z + a.w * b.w;
  }
  acc *= rel_scale[erel[e] * H + h] * 0.25f;      // 1/sqrt(DH)
  attn[t] = acc;
  // block max -> one atomic per block
  float m = acc;
#pragma unroll
  for (int off = 16; off >= 1; off >>= 1) m = fmaxf(m, __shfl_xor(m, off, 32));
  __shared__ float wmax[8];
  if ((threadIdx.x & 31) == 0) wmax[threadIdx.x >> 5] = m;
  __syncthreads();
  if (threadIdx.x == 0) {
    float bm = wmax[0];
#pragma unroll
    for (int i = 1; i < 8; ++i) bm = fmaxf(bm, wmax[i]);
    atomicMax(gmax, f2ord(bm));
  }
}

// ---------------- kernel 4: exp + segment sum over dst ----------------
__global__ void attn_expsum(const int* __restrict__ edst, float* __restrict__ attn,
                            const unsigned* __restrict__ gmax, float* __restrict__ asum) {
  int t = blockIdx.x * 256 + threadIdx.x;         // E*H
  float mv = ord2f(*gmax);
  float ae = __expf(attn[t] - mv);
  attn[t] = ae;
  int d = edst[t >> 3];
  atomAddF(&asum[(size_t)d * H + (t & 7)], ae);
}

// ---------------- kernel 5: normalized message scatter-add ----------------
// 32 lanes per edge; lane owns 4 contiguous channels (one float4).
__global__ void scatter_msg(const int* __restrict__ esrc, const int* __restrict__ edst,
                            const float* __restrict__ attn, const float* __restrict__ asum,
                            const float* __restrict__ Vb, float* __restrict__ agg) {
  int t = blockIdx.x * 256 + threadIdx.x;         // E*32
  int e = t >> 5, lane = t & 31;
  int c0 = lane * 4, h = lane >> 2;               // h = c0/16
  int s = esrc[e], d = edst[e];
  float coeff = attn[(size_t)e * H + h] / (asum[(size_t)d * H + h] + 1e-16f);
  float4 vv = *(const float4*)(Vb + (size_t)s * C + c0);
  float* o = agg + (size_t)d * C + c0;
  atomAddF(o + 0, coeff * vv.x);
  atomAddF(o + 1, coeff * vv.y);
  atomAddF(o + 2, coeff * vv.z);
  atomAddF(o + 3, coeff * vv.w);
}

// ---------------- kernel 6: fused  agg@Wo + bo + skip -> LayerNorm -> ELU ----------------
// One block per 16-row tile; wave w computes cols [16w,16w+16) via WMMA, results to LDS,
// then 16 lanes per row do the LayerNorm reduction in-registers via shuffles.
__global__ void out_ln(const float* __restrict__ agg, const _Float16* __restrict__ WoT,
                       const float* __restrict__ bo, const float* __restrict__ skip,
                       const float* __restrict__ gamma, const float* __restrict__ beta,
                       float* __restrict__ out) {
  __shared__ float hs[16][C + 4];
  const int row0 = blockIdx.x * 16;
  const int wave = threadIdx.x >> 5;
  const int lane = threadIdx.x & 31;
  const float* A = agg + (size_t)row0 * C;
  const _Float16* B = WoT + (size_t)wave * 16 * C;
  v8f c = {};
#pragma unroll
  for (int k0 = 0; k0 < C; k0 += 32) {
    v16h a = load_frag_f32cvt(A, C, k0);
    v16h b = load_frag_f16(B, C, k0);
    c = __builtin_amdgcn_wmma_f32_16x16x32_f16(false, a, false, b, (short)0, c, false, false);
  }
  const int n  = lane & 15;
  const int mb = (lane >> 4) << 3;
  const int col = wave * 16 + n;
  const float bias = bo[col];
#pragma unroll
  for (int v = 0; v < 8; ++v) {
    int m = mb + v;
    hs[m][col] = c[v] + bias + skip[(size_t)(row0 + m) * C + col];
  }
  __syncthreads();
  // LayerNorm + ELU: 16 lanes per row, each lane owns 8 strided channels
  const int row = threadIdx.x >> 4;               // 0..15
  const int sub = threadIdx.x & 15;
  float vals[8], s = 0.f, sq = 0.f;
#pragma unroll
  for (int i = 0; i < 8; ++i) {
    float xv = hs[row][sub + i * 16];
    vals[i] = xv; s += xv; sq += xv * xv;
  }
#pragma unroll
  for (int off = 8; off >= 1; off >>= 1) {        // xor<16 stays within 16-lane half
    s  += __shfl_xor(s,  off, 32);
    sq += __shfl_xor(sq, off, 32);
  }
  const float mu   = s * (1.0f / C);
  const float var  = sq * (1.0f / C) - mu * mu;
  const float rstd = rsqrtf(var + 1e-5f);
#pragma unroll
  for (int i = 0; i < 8; ++i) {
    int cc = sub + i * 16;
    float xv = (vals[i] - mu) * rstd * gamma[cc] + beta[cc];
    out[(size_t)(row0 + row) * C + cc] = (xv > 0.f) ? xv : (__expf(xv) - 1.f);
  }
}

// ---------------- host launcher ----------------
extern "C" void kernel_launch(void* const* d_in, const int* in_sizes, int n_in,
                              void* d_out, int out_size, void* d_ws, size_t ws_size,
                              hipStream_t stream) {
  const float* x         = (const float*)d_in[0];
  const int*   ntype     = (const int*)d_in[1];
  const int*   esrc      = (const int*)d_in[2];
  const int*   edst      = (const int*)d_in[3];
  const int*   erel      = (const int*)d_in[4];
  const float* type_emb  = (const float*)d_in[5];
  const float* Wk        = (const float*)d_in[6];
  const float* Wq        = (const float*)d_in[7];
  const float* Wv        = (const float*)d_in[8];
  const float* rel_scale = (const float*)d_in[9];
  const float* Wo        = (const float*)d_in[10];
  const float* bo        = (const float*)d_in[11];
  const float* Wskip     = (const float*)d_in[12];
  const float* gamma     = (const float*)d_in[13];
  const float* beta      = (const float*)d_in[14];
  float* out = (float*)d_out;

  char* ws = (char*)d_ws;
  size_t off = 0;
  auto take = [&](size_t bytes) { char* p = ws + off; off = (off + bytes + 255) & ~(size_t)255; return p; };
  _Float16* xaug = (_Float16*)take((size_t)N_NODES * CA * 2);
  _Float16* WT1  = (_Float16*)take((size_t)512 * CA * 2);
  _Float16* WoT  = (_Float16*)take((size_t)C * C * 2);
  float* Qb   = (float*)take((size_t)N_NODES * C * 4);
  float* Kb   = (float*)take((size_t)N_NODES * C * 4);
  float* Vb   = (float*)take((size_t)N_NODES * C * 4);
  float* Sb   = (float*)take((size_t)N_NODES * C * 4);
  float* attn = (float*)take((size_t)N_EDGES * H * 4);
  float* asum = (float*)take((size_t)N_NODES * H * 4);
  float* agg  = (float*)take((size_t)N_NODES * C * 4);
  unsigned* gmax = (unsigned*)take(256);

  // per-call re-init (atomically accumulated buffers); 0 == f2ord identity (< f2ord(-inf))
  hipMemsetAsync(asum, 0, (size_t)N_NODES * H * 4, stream);
  hipMemsetAsync(agg,  0, (size_t)N_NODES * C * 4, stream);
  hipMemsetAsync(gmax, 0, 4, stream);

  prep_weights<<<(512 * 256 + C * C) / 256, 256, 0, stream>>>(Wk, Wq, Wv, Wskip, Wo, WT1, WoT);
  build_xaug<<<(N_NODES * CA) / 256, 256, 0, stream>>>(x, ntype, type_emb, xaug);
  dim3 g2(N_NODES / 16, 4);
  gemm_qkvs<<<g2, 256, 0, stream>>>(xaug, WT1, Qb, Kb, Vb, Sb);
  attn_dot<<<(N_EDGES * H) / 256, 256, 0, stream>>>(Qb, Kb, esrc, edst, erel, rel_scale, attn, gmax);
  attn_expsum<<<(N_EDGES * H) / 256, 256, 0, stream>>>(edst, attn, gmax, asum);
  scatter_msg<<<(N_EDGES * 32) / 256, 256, 0, stream>>>(esrc, edst, attn, asum, Vb, agg);
  out_ln<<<N_NODES / 16, 256, 0, stream>>>(agg, WoT, bo, Sb, gamma, beta, out);
}
